// AttentionModel_80822694576536
// MI455X (gfx1250) — matrix operand order repaired and checked by reading, not compiled, not run on hardware
//
#include <hip/hip_runtime.h>
#include <hip/hip_bf16.h>

#define NB   128          // batch
#define LSEQ 256          // sequence length
#define DIM  768          // input dim
#define HID  768          // hidden dim
#define G4H  3072         // 4*H (gates)
#define KCAT 1536         // D + H  (fused LSTM GEMM K)
#define HS_L (LSEQ + 1)   // history slots (slot 0 = zeros)

typedef __attribute__((ext_vector_type(16))) __bf16 v16bf;
typedef __attribute__((ext_vector_type(8)))  __bf16 v8bf;
typedef __attribute__((ext_vector_type(8)))  float  v8f;

// ---------- scalar helpers ----------
__device__ inline __bf16 f2bf(float f) {
  union { float f; unsigned u; } v; v.f = f;
  unsigned r = v.u + 0x7FFFu + ((v.u >> 16) & 1u);   // round-to-nearest-even
  unsigned short h = (unsigned short)(r >> 16);
  __bf16 b; __builtin_memcpy(&b, &h, 2);
  return b;
}
__device__ inline float bf2f(__bf16 b) {
  unsigned short h; __builtin_memcpy(&h, &b, 2);
  union { unsigned u; float f; } v; v.u = ((unsigned)h) << 16;
  return v.f;
}
__device__ inline float sigm(float x) { return 1.0f / (1.0f + expf(-x)); }

// ---------- WMMA fragment loads (wave32, 16x16x32 bf16) ----------
// A (16x32, MxK): lanes 0-15 hold row m=lane, K 0..7 (v0-3) and 16..23 (v4-7);
// lanes 16-31 hold row m=lane-16, K 8..15 and 24..31.
__device__ inline v16bf load_fragA(const __bf16* tile, int ld, int lane) {
  int r  = lane & 15;
  int kh = (lane >> 4) << 3;                    // 0 or 8
  const __bf16* p = tile + (size_t)r * ld + kh;
  v8bf lo = *(const v8bf*)p;                    // K kh..kh+7
  v8bf hi = *(const v8bf*)(p + 16);             // K kh+16..kh+23
  v16bf o;
#pragma unroll
  for (int i = 0; i < 8; ++i) { o[i] = lo[i]; o[i + 8] = hi[i]; }
  return o;
}
// B (32x16, KxN) fed from weights stored N-major (row j of W == column j of B):
// lanes 0-15 hold col n=lane, K 0..7 and 16..23; lanes 16-31 hold col n=lane-16, K 8..15 and 24..31.
__device__ inline v16bf load_fragB(const __bf16* tileT, int ld, int lane) {
  int c  = lane & 15;
  int kh = (lane >> 4) << 3;                    // 0 or 8
  const __bf16* p = tileT + (size_t)c * ld + kh;
  v8bf lo = *(const v8bf*)p;
  v8bf hi = *(const v8bf*)(p + 16);
  v16bf o;
#pragma unroll
  for (int i = 0; i < 8; ++i) { o[i] = lo[i]; o[i + 8] = hi[i]; }
  return o;
}

// ---------- weight / state prep ----------
__global__ void prep_wcat(const float* __restrict__ W_ih,
                          const float* __restrict__ W_hh,
                          __bf16* __restrict__ Wcat) {
  int idx = blockIdx.x * blockDim.x + threadIdx.x;    // over G4H*KCAT
  if (idx >= G4H * KCAT) return;
  int j = idx / KCAT, k = idx % KCAT;
  float v = (k < DIM) ? W_ih[(size_t)j * DIM + k]
                      : W_hh[(size_t)j * HID + (k - DIM)];
  Wcat[idx] = f2bf(v);
}
__global__ void f32_to_bf16_arr(const float* __restrict__ s, __bf16* __restrict__ d, int n) {
  int i = blockIdx.x * blockDim.x + threadIdx.x;
  if (i < n) d[i] = f2bf(s[i]);
}
__global__ void add_bias(const float* a, const float* b, float* o, int n) {
  int i = blockIdx.x * blockDim.x + threadIdx.x;
  if (i < n) o[i] = a[i] + b[i];
}
__global__ void zero_f32(float* p, int n) {
  int i = blockIdx.x * blockDim.x + threadIdx.x;
  if (i < n) p[i] = 0.0f;
}
__global__ void zero_bf16(__bf16* p, size_t n) {
  size_t i = (size_t)blockIdx.x * blockDim.x + threadIdx.x;
  if (i < n) p[i] = f2bf(0.0f);
}

// ---------- per-step kernels ----------
// Build A = [bf16(x_t), bf16(h_tilde)]  (NB x KCAT)
__global__ void prep_A(const float* __restrict__ X, const float* __restrict__ htld,
                       __bf16* __restrict__ A, int t) {
  int idx = blockIdx.x * blockDim.x + threadIdx.x;
  if (idx >= NB * KCAT) return;
  int n = idx / KCAT, k = idx % KCAT;
  float v = (k < DIM) ? X[((size_t)n * LSEQ + t) * DIM + k]
                      : htld[(size_t)n * HID + (k - DIM)];
  A[idx] = f2bf(v);
}

// Fused LSTM cell: gates = A @ Wcat^T + bias; then i,f,g,o -> c,h update.
// One wave per 16(M)x16(Hcol) tile; computes the 4 gate tiles for that Hcol.
__global__ void __launch_bounds__(32)
lstm_gates_kernel(const __bf16* __restrict__ A,      // NB x KCAT
                  const __bf16* __restrict__ Wcat,   // G4H x KCAT (row j = B col j)
                  const float*  __restrict__ bias,   // G4H
                  float* __restrict__ c,             // NB x HID (in/out)
                  float* __restrict__ h,             // NB x HID (out)
                  __bf16* __restrict__ hbf) {        // NB x HID (out, bf16)
  int lane  = threadIdx.x;
  int htile = blockIdx.x;                            // 0..47
  int mtile = blockIdx.y;                            // 0..7
  v8f acc[4] = {};
  for (int k = 0; k < KCAT; k += 32) {
    v16bf a = load_fragA(A + (size_t)mtile * 16 * KCAT + k, KCAT, lane);
#pragma unroll
    for (int g = 0; g < 4; ++g) {
      int j0 = g * HID + htile * 16;
      v16bf b = load_fragB(Wcat + (size_t)j0 * KCAT + k, KCAT, lane);
      acc[g] = __builtin_amdgcn_wmma_f32_16x16x32_bf16(
          false, a, false, b, (short)0, acc[g], false, false);
    }
  }
  int hcol = htile * 16 + (lane & 15);
  float bi = bias[0 * HID + hcol], bf = bias[1 * HID + hcol];
  float bg = bias[2 * HID + hcol], bo = bias[3 * HID + hcol];
#pragma unroll
  for (int r = 0; r < 8; ++r) {
    int m = ((lane >> 4) << 3) + r;
    int n = mtile * 16 + m;
    size_t off = (size_t)n * HID + hcol;
    float gi = sigm(acc[0][r] + bi);
    float gf = sigm(acc[1][r] + bf);
    float gg = tanhf(acc[2][r] + bg);
    float go = sigm(acc[3][r] + bo);
    float cn = gf * c[off] + gi * gg;
    float hn = go * tanhf(cn);
    c[off]   = cn;
    h[off]   = hn;
    hbf[off] = f2bf(hn);
  }
}

// q = h @ att_W^T + att_b
__global__ void __launch_bounds__(32)
att_q_gemm(const __bf16* __restrict__ hbf, const __bf16* __restrict__ attW,
           const float* __restrict__ attb, float* __restrict__ q) {
  int lane = threadIdx.x;
  int ntile = blockIdx.x, mtile = blockIdx.y;
  v8f acc = {};
  for (int k = 0; k < HID; k += 32) {
    v16bf a = load_fragA(hbf + (size_t)mtile * 16 * HID + k, HID, lane);
    v16bf b = load_fragB(attW + (size_t)(ntile * 16) * HID + k, HID, lane);
    acc = __builtin_amdgcn_wmma_f32_16x16x32_bf16(
        false, a, false, b, (short)0, acc, false, false);
  }
  int col = ntile * 16 + (lane & 15);
#pragma unroll
  for (int r = 0; r < 8; ++r) {
    int m = ((lane >> 4) << 3) + r;
    int n = mtile * 16 + m;
    q[(size_t)n * HID + col] = acc[r] + attb[col];
  }
}

// Per-batch attention: scores over history, softmax(valid 0..t), weighted sum.
// Also assembles A_cvt = [h_bf16, h_hat_bf16].
__global__ void __launch_bounds__(256)
attention_kernel(const __bf16* __restrict__ hs, const float* __restrict__ q,
                 const __bf16* __restrict__ hbf, __bf16* __restrict__ Acvt, int t) {
  int n = blockIdx.x, tid = threadIdx.x;
  __shared__ float sq[HID];
  __shared__ float sc[HS_L];
  __shared__ float red[256];
  for (int k = tid; k < HID; k += 256) sq[k] = q[(size_t)n * HID + k];
  __syncthreads();
  int nv = t + 1;                                     // valid positions 0..t
  for (int l = tid; l < nv; l += 256) {
    const __bf16* row = hs + ((size_t)n * HS_L + l) * HID;
    float s = 0.f;
    for (int k = 0; k < HID; ++k) s += bf2f(row[k]) * sq[k];
    sc[l] = s;
  }
  __syncthreads();
  float m = -3.4e38f;
  for (int l = tid; l < nv; l += 256) m = fmaxf(m, sc[l]);
  red[tid] = m; __syncthreads();
  for (int st = 128; st > 0; st >>= 1) {
    if (tid < st) red[tid] = fmaxf(red[tid], red[tid + st]);
    __syncthreads();
  }
  m = red[0]; __syncthreads();
  float ssum = 0.f;
  for (int l = tid; l < nv; l += 256) { float e = expf(sc[l] - m); sc[l] = e; ssum += e; }
  red[tid] = ssum; __syncthreads();
  for (int st = 128; st > 0; st >>= 1) {
    if (tid < st) red[tid] += red[tid + st];
    __syncthreads();
  }
  float inv = 1.0f / red[0];
  __syncthreads();
  for (int l = tid; l < nv; l += 256) sc[l] *= inv;
  __syncthreads();
  for (int k = tid; k < HID; k += 256) {
    float acc = 0.f;
    const __bf16* base = hs + (size_t)n * HS_L * HID + k;
    for (int l = 0; l < nv; ++l) acc += sc[l] * bf2f(base[(size_t)l * HID]);
    Acvt[(size_t)n * KCAT + DIM + k] = f2bf(acc);
    Acvt[(size_t)n * KCAT + k]       = hbf[(size_t)n * HID + k];
  }
}

// h_tilde = A_cvt @ cvt_W^T + cvt_b ; store fp32 state + bf16 history slot t+1.
__global__ void __launch_bounds__(32)
cvt_gemm(const __bf16* __restrict__ Acvt, const __bf16* __restrict__ cvtW,
         const float* __restrict__ cvtb, float* __restrict__ htld,
         __bf16* __restrict__ hs, int t) {
  int lane = threadIdx.x;
  int ntile = blockIdx.x, mtile = blockIdx.y;
  v8f acc = {};
  for (int k = 0; k < KCAT; k += 32) {
    v16bf a = load_fragA(Acvt + (size_t)mtile * 16 * KCAT + k, KCAT, lane);
    v16bf b = load_fragB(cvtW + (size_t)(ntile * 16) * KCAT + k, KCAT, lane);
    acc = __builtin_amdgcn_wmma_f32_16x16x32_bf16(
        false, a, false, b, (short)0, acc, false, false);
  }
  int col = ntile * 16 + (lane & 15);
#pragma unroll
  for (int r = 0; r < 8; ++r) {
    int m = ((lane >> 4) << 3) + r;
    int n = mtile * 16 + m;
    float ht = acc[r] + cvtb[col];
    htld[(size_t)n * HID + col] = ht;
    hs[((size_t)n * HS_L + (t + 1)) * HID + col] = f2bf(ht);
  }
}

// out[n] = sigmoid(h[n] . out_W + out_b)
__global__ void __launch_bounds__(256)
final_out(const float* __restrict__ h, const float* __restrict__ outW,
          const float* __restrict__ outb, float* __restrict__ out) {
  int n = blockIdx.x, tid = threadIdx.x;
  __shared__ float red[256];
  float s = 0.f;
  for (int k = tid; k < HID; k += 256) s += h[(size_t)n * HID + k] * outW[k];
  red[tid] = s; __syncthreads();
  for (int st = 128; st > 0; st >>= 1) {
    if (tid < st) red[tid] += red[tid + st];
    __syncthreads();
  }
  if (tid == 0) out[n] = sigm(red[0] + outb[0]);
}

// ---------- host ----------
extern "C" void kernel_launch(void* const* d_in, const int* in_sizes, int n_in,
                              void* d_out, int out_size, void* d_ws, size_t ws_size,
                              hipStream_t stream) {
  (void)in_sizes; (void)n_in; (void)out_size; (void)ws_size;
  const float* X    = (const float*)d_in[0];
  const float* W_ih = (const float*)d_in[1];
  const float* W_hh = (const float*)d_in[2];
  const float* b_ih = (const float*)d_in[3];
  const float* b_hh = (const float*)d_in[4];
  const float* attW = (const float*)d_in[5];
  const float* attb = (const float*)d_in[6];
  const float* cvtW = (const float*)d_in[7];
  const float* cvtb = (const float*)d_in[8];
  const float* outW = (const float*)d_in[9];
  const float* outb = (const float*)d_in[10];
  float* out = (float*)d_out;

  char* p = (char*)d_ws;
  auto alloc = [&](size_t bytes) {
    char* r = p; p += (bytes + 255) & ~(size_t)255; return r;
  };
  __bf16* Wcat   = (__bf16*)alloc((size_t)G4H * KCAT * 2);   // 9.4 MB
  __bf16* attWbf = (__bf16*)alloc((size_t)HID * HID * 2);    // 1.2 MB
  __bf16* cvtWbf = (__bf16*)alloc((size_t)HID * KCAT * 2);   // 2.4 MB
  float*  bias   = (float*) alloc((size_t)G4H * 4);
  float*  c      = (float*) alloc((size_t)NB * HID * 4);
  float*  h      = (float*) alloc((size_t)NB * HID * 4);
  __bf16* hbf    = (__bf16*)alloc((size_t)NB * HID * 2);
  float*  htld   = (float*) alloc((size_t)NB * HID * 4);
  float*  q      = (float*) alloc((size_t)NB * HID * 4);
  __bf16* Almm   = (__bf16*)alloc((size_t)NB * KCAT * 2);
  __bf16* Acvt   = (__bf16*)alloc((size_t)NB * KCAT * 2);
  __bf16* hs     = (__bf16*)alloc((size_t)NB * HS_L * HID * 2); // 50.5 MB

  // One-time weight conversion + state init
  prep_wcat<<<(G4H * KCAT + 255) / 256, 256, 0, stream>>>(W_ih, W_hh, Wcat);
  f32_to_bf16_arr<<<(HID * HID + 255) / 256, 256, 0, stream>>>(attW, attWbf, HID * HID);
  f32_to_bf16_arr<<<(HID * KCAT + 255) / 256, 256, 0, stream>>>(cvtW, cvtWbf, HID * KCAT);
  add_bias<<<(G4H + 255) / 256, 256, 0, stream>>>(b_ih, b_hh, bias, G4H);
  zero_f32<<<(NB * HID + 255) / 256, 256, 0, stream>>>(c, NB * HID);
  zero_f32<<<(NB * HID + 255) / 256, 256, 0, stream>>>(htld, NB * HID);
  {
    size_t nel = (size_t)NB * HS_L * HID;
    zero_bf16<<<(unsigned)((nel + 255) / 256), 256, 0, stream>>>(hs, nel);
  }

  // Sequential recurrence (graph-captured launches)
  for (int t = 0; t < LSEQ; ++t) {
    prep_A<<<(NB * KCAT + 255) / 256, 256, 0, stream>>>(X, htld, Almm, t);
    lstm_gates_kernel<<<dim3(48, 8), 32, 0, stream>>>(Almm, Wcat, bias, c, h, hbf);
    att_q_gemm<<<dim3(48, 8), 32, 0, stream>>>(hbf, attWbf, attb, q);
    attention_kernel<<<NB, 256, 0, stream>>>(hs, q, hbf, Acvt, t);
    cvt_gemm<<<dim3(48, 8), 32, 0, stream>>>(Acvt, cvtWbf, cvtb, htld, hs, t);
  }
  final_out<<<NB, 256, 0, stream>>>(h, outW, outb, out);
}
